// SignLLM_VQ_67697274519870
// MI455X (gfx1250) — matrix operand order repaired and checked by reading, not compiled, not run on hardware
//
#include <hip/hip_runtime.h>

// ============================================================================
// MI455X (gfx1250, wave32) VQ-transformer forward.
//   - bf16 WMMA (v_wmma_f32_16x16x32_bf16) for all matmuls, fp32 accumulate.
//   - weights converted once to bf16 (~28MB -> L2-resident on 192MB L2).
//   - GEMM tiles staged by the Tensor Data Mover (tensor_load_to_lds, one
//     issuing wave, s_wait_tensorcnt + barrier), double-buffered so the DMA
//     overlaps the WMMA pipe; waves issue only ds_load fragments + WMMA.
//   - fused flash-attention (avoids ~1.6GB score/prob HBM traffic ~ 70us
//     @23.3TB/s) with pure-VALU DPP butterfly reductions for online softmax.
// ============================================================================

typedef unsigned short u16;
typedef unsigned int   u32;
typedef __attribute__((ext_vector_type(16))) __bf16 v16bf;
typedef __attribute__((ext_vector_type(8)))  __bf16 v8bf;
typedef __attribute__((ext_vector_type(8)))  float  v8f;
typedef __attribute__((ext_vector_type(4)))  unsigned int v4u;
typedef __attribute__((ext_vector_type(8)))  int v8i;
typedef __attribute__((ext_vector_type(4)))  int v4i;

// ---------------- helpers ----------------
__device__ inline u16 f2bf(float f) {             // RNE fp32 -> bf16
    u32 u = __builtin_bit_cast(u32, f);
    u32 r = u + 0x7FFFu + ((u >> 16) & 1u);
    return (u16)(r >> 16);
}

__device__ inline v16bf mk16(uint4 c0, uint4 c1) {
    v8bf lo = __builtin_bit_cast(v8bf, c0);
    v8bf hi = __builtin_bit_cast(v8bf, c1);
    v16bf r;
#pragma unroll
    for (int i = 0; i < 8; i++) { r[i] = lo[i]; r[i + 8] = hi[i]; }
    return r;
}

// A-fragment (16xK tile, 16-bit): lane m = l%16; chunks at k = g*8 and g*8+16
__device__ inline v16bf frag_a(const u16* sm, int row0, int ldk, int koff, int lane) {
    const u16* p = sm + (size_t)(row0 + (lane & 15)) * ldk + koff + (lane >> 4) * 8;
    return mk16(*(const uint4*)p, *(const uint4*)(p + 16));
}

// B-fragment (Kx16 tile stored [n][k], 16-bit): lane n = l%16; chunks at k = g*8 and g*8+16
__device__ inline v16bf frag_b(const u16* sm, int col0, int ldk, int koff, int lane) {
    const u16* p = sm + (size_t)(col0 + (lane & 15)) * ldk + koff + (lane >> 4) * 8;
    return mk16(*(const uint4*)p, *(const uint4*)(p + 16));
}

__device__ inline v8f wmma_bf16(v16bf a, v16bf b, v8f c) {
    return __builtin_amdgcn_wmma_f32_16x16x32_bf16(false, a, false, b, (short)0, c,
                                                   false, false);
}

// ---- Tensor Data Mover: 2-D tile (rows x 32 bf16, row stride `ld`) -> LDS ----
// D# built per CDNA5 ISA 8.3/8.4: group0 {count=1, lds_addr, 57b global addr,
// type=2}; group1 {data_size=2B, big tensor dims (tile always in-bounds),
// tile_dim0=32, tile_dim1=rows, dim0 stride}; trailing groups zero => 2-D.
// This toolchain exposes the 6-arg builtin form (clang-23 / therock-10.0).
__device__ inline void tdm_load_tile(const u16* gaddr, u32 ldsOff, int rows, int ld) {
    unsigned long long ga = (unsigned long long)(size_t)gaddr;
    v4u g0;
    g0[0] = 1u;                                        // count=1, user descriptor
    g0[1] = ldsOff;                                    // lds_addr (bytes)
    g0[2] = (u32)ga;                                   // global_addr[31:0]
    g0[3] = (u32)((ga >> 32) & 0x1FFFFFFu) | (2u << 30); // addr[56:32] | type=2
    const u32 td0 = 1u << 20, td1 = 1u << 20;          // OOB limits (never hit)
    v8i g1;
    g1[0] = (int)(1u << 16);                           // data_size code 1 = 2B
    g1[1] = (int)((td0 & 0xFFFFu) << 16);              // tensor_dim0[15:0]
    g1[2] = (int)((td0 >> 16) | ((td1 & 0xFFFFu) << 16));
    g1[3] = (int)((td1 >> 16) | (32u << 16));          // tile_dim0 = 32
    g1[4] = rows;                                      // tile_dim1; tile_dim2=0
    g1[5] = ld;                                        // tensor_dim0_stride[31:0]
    g1[6] = 0;                                         // stride hi; dim1_stride=0
    g1[7] = 0;
    v4i z4 = {};
    v8i z8 = {};
    __builtin_amdgcn_tensor_load_to_lds(g0, g1, z4, z4, z8, 0);
}

__device__ inline u32 lds_off(const void* p) {        // generic ptr -> LDS offset
    return (u32)(size_t)p;                            // LDS aperture: addr[31:0]
}

// ---- DPP butterfly reductions over 16-lane groups (pure VALU, no LDS) ----
#define DPP_PERM(x, ctrl) __builtin_amdgcn_update_dpp(0, (x), (ctrl), 0xf, 0xf, true)

__device__ inline float red16_max(float v) {
    v = fmaxf(v, __builtin_bit_cast(float, DPP_PERM(__builtin_bit_cast(int, v), 0xB1)));
    v = fmaxf(v, __builtin_bit_cast(float, DPP_PERM(__builtin_bit_cast(int, v), 0x4E)));
    v = fmaxf(v, __builtin_bit_cast(float, DPP_PERM(__builtin_bit_cast(int, v), 0x141)));
    v = fmaxf(v, __builtin_bit_cast(float, DPP_PERM(__builtin_bit_cast(int, v), 0x140)));
    return v;
}
__device__ inline float red16_sum(float v) {
    v += __builtin_bit_cast(float, DPP_PERM(__builtin_bit_cast(int, v), 0xB1));
    v += __builtin_bit_cast(float, DPP_PERM(__builtin_bit_cast(int, v), 0x4E));
    v += __builtin_bit_cast(float, DPP_PERM(__builtin_bit_cast(int, v), 0x141));
    v += __builtin_bit_cast(float, DPP_PERM(__builtin_bit_cast(int, v), 0x140));
    return v;
}

__device__ inline float block_sum(float v, float* red) {   // 256 threads
    int t = threadIdx.x;
    red[t] = v; __syncthreads();
#pragma unroll
    for (int s = 128; s > 0; s >>= 1) {
        if (t < s) red[t] += red[t + s];
        __syncthreads();
    }
    float r = red[0]; __syncthreads();
    return r;
}

// ---------------- generic WMMA GEMM: C = act(A @ W^T + bias (+res)) --------
// A: bf16 [M,K] (lda), W: bf16 [N,K] (ldb).  Block tile 128x64, 8 waves
// (4 in M x 2 in N), each wave 32x32 = 4 accumulators. K-step 32. Tiles are
// DMA'd by the TDM (wave 0 issues, overlapping compute), LDS double-buffered.
// M%128==0, N%64==0, K%32==0 (true for all shapes in this model).
template <bool HAS_BIAS, bool RELU, bool HAS_RES, bool OUT_F32, bool OUT_BF16>
__global__ __launch_bounds__(256) void gemm_k(
    const u16* __restrict__ A, int lda,
    const u16* __restrict__ Bw, int ldb,
    float* __restrict__ Cf, u16* __restrict__ Cb, int ldc,
    const float* __restrict__ bias,
    const float* __restrict__ res,
    int Kdim)
{
    __shared__ __align__(16) u16 As[2][128 * 32];
    __shared__ __align__(16) u16 Bs[2][64 * 32];
    const int tid = threadIdx.x, wave = tid >> 5, lane = tid & 31;
    const int m0 = blockIdx.y * 128, n0 = blockIdx.x * 64;
    const int wm = (wave & 3) * 32, wn = (wave >> 2) * 32;
    v8f acc00 = {}, acc01 = {}, acc10 = {}, acc11 = {};

    const u16* aTile = A  + (size_t)m0 * lda;      // +k0 walks the K dimension
    const u16* bTile = Bw + (size_t)n0 * ldb;

    auto compute = [&](int bsel) {
        const u16* Ap = As[bsel];
        const u16* Bp = Bs[bsel];
        v16bf a0 = frag_a(Ap, wm,      32, 0, lane);
        v16bf a1 = frag_a(Ap, wm + 16, 32, 0, lane);
        v16bf b0 = frag_b(Bp, wn,      32, 0, lane);
        v16bf b1 = frag_b(Bp, wn + 16, 32, 0, lane);
        acc00 = wmma_bf16(a0, b0, acc00);
        acc01 = wmma_bf16(a0, b1, acc01);
        acc10 = wmma_bf16(a1, b0, acc10);
        acc11 = wmma_bf16(a1, b1, acc11);
    };

    // prologue: TDM tile 0 into buffer 0
    if (wave == 0) {
        tdm_load_tile(aTile, lds_off(&As[0][0]), 128, lda);
        tdm_load_tile(bTile, lds_off(&Bs[0][0]),  64, ldb);
        __builtin_amdgcn_s_wait_tensorcnt(0);
    }
    __syncthreads();

    int buf = 0;
    for (int k0 = 32; k0 < Kdim; k0 += 32) {
        if (wave == 0) {                               // DMA next tile (overlaps)
            tdm_load_tile(aTile + k0, lds_off(&As[buf ^ 1][0]), 128, lda);
            tdm_load_tile(bTile + k0, lds_off(&Bs[buf ^ 1][0]),  64, ldb);
        }
        compute(buf);                                  // WMMAs overlap the DMA
        if (wave == 0) __builtin_amdgcn_s_wait_tensorcnt(0);
        __syncthreads();
        buf ^= 1;
    }
    compute(buf);

    const int g = lane >> 4, cn = lane & 15;
    auto epi = [&](const v8f& a, int mtOff, int ntOff) {
        const int row0 = m0 + wm + mtOff + 8 * g;
        const int col  = n0 + wn + ntOff + cn;
#pragma unroll
        for (int r = 0; r < 8; r++) {
            const int row = row0 + r;
            float v = a[r];
            if (HAS_BIAS) v += bias[col];
            if (HAS_RES)  v += res[(size_t)row * ldc + col];
            if (RELU)     v = fmaxf(v, 0.f);
            if (OUT_F32)  Cf[(size_t)row * ldc + col] = v;
            if (OUT_BF16) Cb[(size_t)row * ldc + col] = f2bf(v);
        }
    };
    epi(acc00, 0, 0); epi(acc01, 0, 16); epi(acc10, 16, 0); epi(acc11, 16, 16);
}

// ---------------- fused flash attention (S=512, H=8, DH=64) ----------------
// grid = (S/64 q-tiles, B*H). block = 128 = 4 waves, each wave owns 16 q-rows.
__global__ __launch_bounds__(128) void attn_k(
    const u16* __restrict__ qkv,   // bf16 [B*S, 1536]
    u16* __restrict__ obuf,        // bf16 [B*S, 512]
    float scale)
{
    const int S = 512, LDQ = 1536, CDM = 512;
    const int bh = blockIdx.y, b = bh >> 3, h = bh & 7;
    const int q0 = blockIdx.x * 64;
    const int tid = threadIdx.x, wave = tid >> 5, lane = tid & 31;
    const u16* base = qkv + (size_t)b * S * LDQ + h * 64;
    const u16* qp = base;
    const u16* kp = base + 512;
    const u16* vp = base + 1024;

    __shared__ __align__(16) u16 Ks[32 * 64];      // [key][feat]
    __shared__ __align__(16) u16 Vt[64 * 32];      // [feat][key] (transposed)
    __shared__ __align__(16) u16 Ps[4][16 * 32];   // per-wave P staging

    v16bf qf0, qf1;
    {
        const u16* p = qp + (size_t)(q0 + wave * 16 + (lane & 15)) * LDQ + (lane >> 4) * 8;
        qf0 = mk16(*(const uint4*)p,        *(const uint4*)(p + 16));
        qf1 = mk16(*(const uint4*)(p + 32), *(const uint4*)(p + 48));
    }

    v8f o0 = {}, o1 = {}, o2 = {}, o3 = {};
    float mx[8], sd[8];
#pragma unroll
    for (int r = 0; r < 8; r++) { mx[r] = -3.0e38f; sd[r] = 0.f; }

    const int skey = tid >> 2, sc = (tid & 3) * 16;
    for (int kb = 0; kb < S; kb += 32) {
        {   // stage K tile (contiguous rows)
            const u16* src = kp + (size_t)(kb + skey) * LDQ + sc;
            *(uint4*)&Ks[skey * 64 + sc]     = *(const uint4*)src;
            *(uint4*)&Ks[skey * 64 + sc + 8] = *(const uint4*)(src + 8);
        }
        {   // stage V tile transposed -> [feat][key]
            const u16* src = vp + (size_t)(kb + skey) * LDQ + sc;
            uint4 d0 = *(const uint4*)src, d1 = *(const uint4*)(src + 8);
            u16 tmp[16];
            *(uint4*)&tmp[0] = d0; *(uint4*)&tmp[8] = d1;
#pragma unroll
            for (int j = 0; j < 16; j++) Vt[(sc + j) * 32 + skey] = tmp[j];
        }
        __syncthreads();

        // scores for 2 key sub-tiles, K-dim 64 = 2 WMMA steps
        v8f s0 = {}, s1 = {};
        s0 = wmma_bf16(qf0, frag_b(Ks, 0, 64, 0, lane), s0);
        s0 = wmma_bf16(qf1, frag_b(Ks, 0, 64, 32, lane), s0);
        s1 = wmma_bf16(qf0, frag_b(Ks, 16, 64, 0, lane), s1);
        s1 = wmma_bf16(qf1, frag_b(Ks, 16, 64, 32, lane), s1);

        float p0[8], p1[8], fr[8];
#pragma unroll
        for (int r = 0; r < 8; r++) {
            float a = s0[r] * scale, c = s1[r] * scale;
            float v = red16_max(fmaxf(a, c));           // DPP butterfly (VALU)
            float nm = fmaxf(mx[r], v);
            fr[r] = __expf(mx[r] - nm);
            p0[r] = __expf(a - nm);
            p1[r] = __expf(c - nm);
            float ps = red16_sum(p0[r] + p1[r]);        // DPP butterfly (VALU)
            sd[r] = sd[r] * fr[r] + ps;
            mx[r] = nm;
        }
#pragma unroll
        for (int r = 0; r < 8; r++) {
            o0[r] *= fr[r]; o1[r] *= fr[r]; o2[r] *= fr[r]; o3[r] *= fr[r];
        }

        // C-layout -> A-layout transpose of P through per-wave LDS
        {
            u16* pw = &Ps[wave][0];
            const int prow = (lane >> 4) * 8, pcol = lane & 15;
#pragma unroll
            for (int r = 0; r < 8; r++) {
                pw[(prow + r) * 32 + pcol]      = f2bf(p0[r]);
                pw[(prow + r) * 32 + 16 + pcol] = f2bf(p1[r]);
            }
        }
        v16bf pf = frag_a(&Ps[wave][0], 0, 32, 0, lane);
        o0 = wmma_bf16(pf, frag_b(Vt, 0, 32, 0, lane), o0);
        o1 = wmma_bf16(pf, frag_b(Vt, 16, 32, 0, lane), o1);
        o2 = wmma_bf16(pf, frag_b(Vt, 32, 32, 0, lane), o2);
        o3 = wmma_bf16(pf, frag_b(Vt, 48, 32, 0, lane), o3);
        __syncthreads();
    }

    const int g = lane >> 4, cn = lane & 15;
#pragma unroll
    for (int r = 0; r < 8; r++) {
        const float inv = 1.f / sd[r];
        const size_t row = (size_t)b * S + q0 + wave * 16 + 8 * g + r;
        const size_t col = (size_t)h * 64 + cn;
        obuf[row * CDM + col]      = f2bf(o0[r] * inv);
        obuf[row * CDM + col + 16] = f2bf(o1[r] * inv);
        obuf[row * CDM + col + 32] = f2bf(o2[r] * inv);
        obuf[row * CDM + col + 48] = f2bf(o3[r] * inv);
    }
}

// ---------------- LayerNorm (one block of 256 per row) ----------------------
template <bool RELU, bool OUT_F32, bool OUT_BF16>
__global__ __launch_bounds__(256) void ln_k(
    const float* __restrict__ in, const float* __restrict__ gam,
    const float* __restrict__ bet, float* __restrict__ outf,
    u16* __restrict__ outb, int D)
{
    __shared__ float red[256];
    const size_t row = blockIdx.x;
    const float* x = in + row * D;
    float s = 0.f;
    for (int i = threadIdx.x; i < D; i += 256) s += x[i];
    s = block_sum(s, red);
    const float mean = s / D;
    float v = 0.f;
    for (int i = threadIdx.x; i < D; i += 256) { float d = x[i] - mean; v += d * d; }
    v = block_sum(v, red);
    const float rstd = rsqrtf(v / D + 1e-5f);
    for (int i = threadIdx.x; i < D; i += 256) {
        float y = (x[i] - mean) * rstd * gam[i] + bet[i];
        if (RELU) y = fmaxf(y, 0.f);
        if (OUT_F32)  outf[row * D + i] = y;
        if (OUT_BF16) outb[row * D + i] = f2bf(y);
    }
}

// ---------------- small kernels ---------------------------------------------
__global__ __launch_bounds__(256) void cvt_k(const float* __restrict__ in,
                                             u16* __restrict__ out, long n)
{
    for (long i = (long)blockIdx.x * 256 + threadIdx.x; i < n; i += (long)gridDim.x * 256)
        out[i] = f2bf(in[i]);
}

__global__ __launch_bounds__(256) void cnorm_k(const float* __restrict__ cb,
                                               u16* __restrict__ cb_bf,
                                               float* __restrict__ cn)
{
    __shared__ float red[256];
    const size_t row = blockIdx.x;
    float s = 0.f;
    for (int i = threadIdx.x; i < 512; i += 256) {
        float c = cb[row * 512 + i];
        s += c * c;
        cb_bf[row * 512 + i] = f2bf(c);
    }
    s = block_sum(s, red);
    if (threadIdx.x == 0) cn[row] = s;
}

// argmin over j of (|c_j|^2 - 2 z.c_j)  (|z|^2 is row-constant); first-min ties
__global__ __launch_bounds__(256) void argmin_k(const float* __restrict__ scores,
                                                const float* __restrict__ cnorm,
                                                int* __restrict__ tok,
                                                int* __restrict__ outTok)
{
    __shared__ float bv[256];
    __shared__ int   bi[256];
    const size_t row = blockIdx.x;
    const float* sr = scores + row * 1024;
    float best = 3.0e38f; int bidx = 0;
    for (int j = threadIdx.x; j < 1024; j += 256) {
        float v = cnorm[j] - 2.f * sr[j];
        if (v < best || (v == best && j < bidx)) { best = v; bidx = j; }
    }
    bv[threadIdx.x] = best; bi[threadIdx.x] = bidx; __syncthreads();
#pragma unroll
    for (int s = 128; s > 0; s >>= 1) {
        if (threadIdx.x < s) {
            float ov = bv[threadIdx.x + s]; int oi = bi[threadIdx.x + s];
            if (ov < bv[threadIdx.x] || (ov == bv[threadIdx.x] && oi < bi[threadIdx.x])) {
                bv[threadIdx.x] = ov; bi[threadIdx.x] = oi;
            }
        }
        __syncthreads();
    }
    if (threadIdx.x == 0) { tok[row] = bi[0]; outTok[row] = bi[0]; }
}

__global__ __launch_bounds__(256) void gather_k(const float* __restrict__ cb,
                                                const int* __restrict__ tok,
                                                float* __restrict__ zqf,
                                                u16* __restrict__ zqb)
{
    const size_t row = blockIdx.x;
    const int t = tok[row];
    for (int i = threadIdx.x; i < 512; i += 256) {
        float v = cb[(size_t)t * 512 + i];
        zqf[row * 512 + i] = v;
        zqb[row * 512 + i] = f2bf(v);
    }
}

__global__ __launch_bounds__(256) void sqsum_k(const float* __restrict__ a,
                                               const float* __restrict__ b,
                                               long n, float* __restrict__ acc)
{
    __shared__ float red[256];
    float s = 0.f;
    for (long i = (long)blockIdx.x * 256 + threadIdx.x; i < n; i += (long)gridDim.x * 256) {
        float d = a[i] - b[i];
        s += d * d;
    }
    s = block_sum(s, red);
    if (threadIdx.x == 0) atomicAdd(acc, s);
}

__global__ void zero2_k(float* a) { a[0] = 0.f; a[1] = 0.f; }

__global__ void final_k(const float* s, float* out, float inv1, float inv2)
{
    float recon = s[0] * inv1;
    float commit = s[1] * inv2;               // commit == codebook loss in fwd
    out[0] = recon + 1.25f * commit;          // recon + commit + 0.25*codebook
    out[1] = recon;
}

// ============================================================================
extern "C" void kernel_launch(void* const* d_in, const int* in_sizes, int n_in,
                              void* d_out, int out_size, void* d_ws, size_t ws_size,
                              hipStream_t stream)
{
    (void)in_sizes; (void)n_in; (void)out_size; (void)ws_size;
    constexpr int BB = 16, SS = 512, NT = BB * SS;     // 8192 tokens
    constexpr int IN = 256, CDM = 512, EH = 768, NC = 1024, L = 4, FFD = 2048;

    const float* x    = (const float*)d_in[0];
    const float* We1  = (const float*)d_in[1];
    const float* be1  = (const float*)d_in[2];
    const float* eg   = (const float*)d_in[3];
    const float* eb   = (const float*)d_in[4];
    const float* We2  = (const float*)d_in[5];
    const float* be2  = (const float*)d_in[6];
    const float* cb   = (const float*)d_in[7];
    const float* Wqkv = (const float*)d_in[8];
    const float* bqkv = (const float*)d_in[9];
    const float* Wo   = (const float*)d_in[10];
    const float* bo   = (const float*)d_in[11];
    const float* ln1g = (const float*)d_in[12];
    const float* ln1b = (const float*)d_in[13];
    const float* W1   = (const float*)d_in[14];
    const float* b1   = (const float*)d_in[15];
    const float* W2   = (const float*)d_in[16];
    const float* b2   = (const float*)d_in[17];
    const float* ln2g = (const float*)d_in[18];
    const float* ln2b = (const float*)d_in[19];
    const float* Wd   = (const float*)d_in[20];
    const float* bd   = (const float*)d_in[21];

    float* outF   = (float*)d_out;
    int*   outTok = (int*)d_out + 2;

    // ---- workspace carving (aliased scratch regions) ----
    unsigned char* w = (unsigned char*)d_ws;
    size_t off = 0;
    auto carve = [&](size_t bytes) -> void* {
        void* p = w + off;
        off += (bytes + 255) & ~(size_t)255;
        return p;
    };
    u16*   x_bf  = (u16*)  carve((size_t)NT * IN * 2);
    u16*   We1b  = (u16*)  carve((size_t)EH * IN * 2);
    u16*   We2b  = (u16*)  carve((size_t)CDM * EH * 2);
    u16*   Wqkvb = (u16*)  carve((size_t)L * 3 * CDM * CDM * 2);
    u16*   Wob   = (u16*)  carve((size_t)L * CDM * CDM * 2);
    u16*   W1b   = (u16*)  carve((size_t)L * FFD * CDM * 2);
    u16*   W2b   = (u16*)  carve((size_t)L * CDM * FFD * 2);
    u16*   Wdb   = (u16*)  carve((size_t)IN * CDM * 2);
    u16*   cbb   = (u16*)  carve((size_t)NC * CDM * 2);
    float* cnorm = (float*)carve((size_t)NC * 4);
    float* ze_f  = (float*)carve((size_t)NT * CDM * 4);
    u16*   ze_b  = (u16*)  carve((size_t)NT * CDM * 2);
    float* zq_f  = (float*)carve((size_t)NT * CDM * 4);
    u16*   zq_b  = (u16*)  carve((size_t)NT * CDM * 2);
    int*   tok   = (int*)  carve((size_t)NT * 4);
    float* ctx_f = (float*)carve((size_t)NT * CDM * 4);
    u16*   ctx_b = (u16*)  carve((size_t)NT * CDM * 2);
    float* sum_f = (float*)carve((size_t)NT * CDM * 4);
    float* lsum  = (float*)carve(256);
    void*  scr1  = carve((size_t)NT * NC * 4);   // scores  | ff_bf
    void*  scr2  = carve((size_t)NT * EH * 4);   // hpre    | qkv_bf
    void*  scr3  = carve((size_t)NT * EH * 2);   // h_bf    | o_bf | recon
    float* scores = (float*)scr1;  u16* ff_b  = (u16*)scr1;
    float* hpre   = (float*)scr2;  u16* qkv_b = (u16*)scr2;
    u16*   h_bf   = (u16*)scr3;    u16* o_bf  = (u16*)scr3;  float* recon = (float*)scr3;

    auto cvt = [&](const float* src, u16* dst, long n) {
        int blocks = (int)((n + 255) / 256);
        if (blocks > 4096) blocks = 4096;
        cvt_k<<<blocks, 256, 0, stream>>>(src, dst, n);
    };

    // ---- fp32 -> bf16 conversions (weights L2-resident afterwards) ----
    cvt(x,    x_bf,  (long)NT * IN);
    cvt(We1,  We1b,  (long)EH * IN);
    cvt(We2,  We2b,  (long)CDM * EH);
    cvt(Wqkv, Wqkvb, (long)L * 3 * CDM * CDM);
    cvt(Wo,   Wob,   (long)L * CDM * CDM);
    cvt(W1,   W1b,   (long)L * FFD * CDM);
    cvt(W2,   W2b,   (long)L * CDM * FFD);
    cvt(Wd,   Wdb,   (long)IN * CDM);
    cnorm_k<<<NC, 256, 0, stream>>>(cb, cbb, cnorm);

    // ---- encoder MLP ----
    gemm_k<true, false, false, true, false><<<dim3(EH / 64, NT / 128), 256, 0, stream>>>(
        x_bf, IN, We1b, IN, hpre, nullptr, EH, be1, nullptr, IN);
    ln_k<true, false, true><<<NT, 256, 0, stream>>>(hpre, eg, eb, nullptr, h_bf, EH);
    gemm_k<true, false, false, true, true><<<dim3(CDM / 64, NT / 128), 256, 0, stream>>>(
        h_bf, EH, We2b, EH, ze_f, ze_b, CDM, be2, nullptr, EH);

    // ---- VQ: z @ cb^T GEMM, argmin(cnorm - 2*score), gather ----
    gemm_k<false, false, false, true, false><<<dim3(NC / 64, NT / 128), 256, 0, stream>>>(
        ze_b, CDM, cbb, CDM, scores, nullptr, NC, nullptr, nullptr, CDM);
    argmin_k<<<NT, 256, 0, stream>>>(scores, cnorm, tok, outTok);
    gather_k<<<NT, 256, 0, stream>>>(cb, tok, zq_f, zq_b);

    // ---- transformer layers (post-norm) ----
    const u16*  in_b = zq_b;
    const float* in_f = zq_f;
    for (int l = 0; l < L; l++) {
        gemm_k<true, false, false, false, true><<<dim3(3 * CDM / 64, NT / 128), 256, 0, stream>>>(
            in_b, CDM, Wqkvb + (size_t)l * 3 * CDM * CDM, CDM,
            nullptr, qkv_b, 3 * CDM, bqkv + (size_t)l * 3 * CDM, nullptr, CDM);
        attn_k<<<dim3(SS / 64, BB * 8), 128, 0, stream>>>(qkv_b, o_bf, 0.125f);
        gemm_k<true, false, true, true, false><<<dim3(CDM / 64, NT / 128), 256, 0, stream>>>(
            o_bf, CDM, Wob + (size_t)l * CDM * CDM, CDM,
            sum_f, nullptr, CDM, bo + (size_t)l * CDM, in_f, CDM);
        ln_k<false, true, true><<<NT, 256, 0, stream>>>(
            sum_f, ln1g + (size_t)l * CDM, ln1b + (size_t)l * CDM, ctx_f, ctx_b, CDM);
        gemm_k<true, true, false, false, true><<<dim3(FFD / 64, NT / 128), 256, 0, stream>>>(
            ctx_b, CDM, W1b + (size_t)l * FFD * CDM, CDM,
            nullptr, ff_b, FFD, b1 + (size_t)l * FFD, nullptr, CDM);
        gemm_k<true, false, true, true, false><<<dim3(CDM / 64, NT / 128), 256, 0, stream>>>(
            ff_b, FFD, W2b + (size_t)l * CDM * FFD, FFD,
            sum_f, nullptr, CDM, b2 + (size_t)l * CDM, ctx_f, FFD);
        ln_k<false, true, true><<<NT, 256, 0, stream>>>(
            sum_f, ln2g + (size_t)l * CDM, ln2b + (size_t)l * CDM, ctx_f, ctx_b, CDM);
        in_b = ctx_b;
        in_f = ctx_f;
    }

    // ---- decoder + losses ----
    gemm_k<true, false, false, true, false><<<dim3(IN / 64, NT / 128), 256, 0, stream>>>(
        ctx_b, CDM, Wdb, CDM, recon, nullptr, IN, bd, nullptr, CDM);
    zero2_k<<<1, 1, 0, stream>>>(lsum);
    sqsum_k<<<1024, 256, 0, stream>>>(recon, x, (long)NT * IN, lsum);
    sqsum_k<<<2048, 256, 0, stream>>>(ze_f, zq_f, (long)NT * CDM, lsum + 1);
    final_k<<<1, 1, 0, stream>>>(lsum, outF,
                                 1.f / ((float)NT * IN), 1.f / ((float)NT * CDM));
}
